// TemplatePointwiseAttention_64707977281622
// MI455X (gfx1250) — matrix operand order repaired and checked by reading, not compiled, not run on hardware
//
#include <hip/hip_runtime.h>
#include <stdint.h>

// ---------------------------------------------------------------------------
// Template pointwise attention (AlphaFold) for gfx1250 / MI455X.
// bf16 WMMA (v_wmma_f32_16x16x32_bf16) for all four GEMMs, VALU for the
// tiny T=4 softmax. Single pass over t/z (memory-bound target ~23us).
// v_perm_b32 for bf16 pair packing; global_prefetch_b8 for next-tile data.
// ---------------------------------------------------------------------------

typedef __bf16 v16bf __attribute__((ext_vector_type(16)));
typedef float  v8f   __attribute__((ext_vector_type(8)));

#define TPA_T   4
#define TPA_N   512
#define TPA_NP  (TPA_N * TPA_N)      // 262144 pairs
#define TPA_CT  64
#define TPA_CZ  128
#define TILES   (TPA_NP / 16)        // 16384
#define WAVES   4
#define PITCH_H 72                   // bf16 elems per staged row (64 + 8 pad)
#define OPITCH  36                   // u32 per o row (32 + 4 pad)

union Frag {
    v16bf    v;
    uint16_t s[16];
    uint32_t u[8];
};

__device__ __forceinline__ uint16_t f2bf_rne(float x) {          // round-nearest-even
    uint32_t u = __float_as_uint(x);
    return (uint16_t)((u + 0x7FFFu + ((u >> 16) & 1u)) >> 16);
}
__device__ __forceinline__ uint16_t f2bf(float x) {              // cheap round
    return (uint16_t)((__float_as_uint(x) + 0x8000u) >> 16);
}

// pack two f32 -> {bf16(hi)<<16 | bf16(lo)} : 2x v_add + 1x v_perm_b32
__device__ __forceinline__ uint32_t pack2bf(float lo, float hi) {
    uint32_t a = __float_as_uint(lo) + 0x8000u;
    uint32_t b = __float_as_uint(hi) + 0x8000u;
    // byte vector = {S0=b (bytes 7..4), S1=a (bytes 3..0)}; pick a[3:2], b[3:2]
    return __builtin_amdgcn_perm(b, a, 0x07060302u);
}

__device__ __forceinline__ v8f wmma_bf16(const Frag& a, const Frag& b, v8f c) {
    // D = A(16x32 bf16) x B(32x16 bf16) + C(16x16 f32)
    return __builtin_amdgcn_wmma_f32_16x16x32_bf16(
        /*neg_a=*/false, a.v, /*neg_b=*/false, b.v,
        /*c_mod=*/(short)0, c, /*reuse_a=*/false, /*reuse_b=*/false);
}

// B fragment from pre-swizzled LDS weights: frag f, 32B per lane.
__device__ __forceinline__ Frag load_bfrag(const uint32_t* s_w, int f, int lane) {
    Frag r;
    const uint4* p = (const uint4*)(s_w + f * 256 + lane * 8);
    uint4 a = p[0], b = p[1];
    r.u[0] = a.x; r.u[1] = a.y; r.u[2] = a.z; r.u[3] = a.w;
    r.u[4] = b.x; r.u[5] = b.y; r.u[6] = b.z; r.u[7] = b.w;
    return r;
}

// A fragment (16x32 bf16) from a global f32 row. Per 16-bit A layout:
// halves 0..7 = K kb..kb+7, halves 8..15 = K kb+16..kb+23 (kb already
// includes the +8 for lanes 16..31).
__device__ __forceinline__ Frag load_afrag_global(const float* row, int kb) {
    Frag r;
    const float4* p0 = (const float4*)(row + kb);
    const float4* p1 = (const float4*)(row + kb + 16);
    float4 x0 = p0[0], x1 = p0[1], y0 = p1[0], y1 = p1[1];
    float f[16] = {x0.x,x0.y,x0.z,x0.w, x1.x,x1.y,x1.z,x1.w,
                   y0.x,y0.y,y0.z,y0.w, y1.x,y1.y,y1.z,y1.w};
    #pragma unroll
    for (int i = 0; i < 8; ++i) r.u[i] = pack2bf(f[2*i], f[2*i+1]);
    return r;
}

// 16 consecutive staged bf16 -> f32
__device__ __forceinline__ void load16bf(const uint16_t* p, float* f) {
    const uint4* q = (const uint4*)p;
    uint4 a = q[0], b = q[1];
    uint32_t w[8] = {a.x,a.y,a.z,a.w, b.x,b.y,b.z,b.w};
    #pragma unroll
    for (int i = 0; i < 8; ++i) {
        f[2*i]   = __uint_as_float(w[i] << 16);
        f[2*i+1] = __uint_as_float(w[i] & 0xFFFF0000u);
    }
}

__global__ __launch_bounds__(WAVES * 32)
void tpa_wmma_kernel(const float* __restrict__ tmpl,   // [T,N,N,64]
                     const float* __restrict__ z,      // [N,N,128]
                     const float* __restrict__ tmask,  // [T]
                     const float* __restrict__ Wq,     // [128,64]
                     const float* __restrict__ Wk,     // [64,64]
                     const float* __restrict__ Wv,     // [64,64]
                     const float* __restrict__ Wo,     // [64,128]
                     const float* __restrict__ bo,     // [128]
                     float* __restrict__ out)          // [N,N,128]
{
    // Pre-swizzled bf16 weights: 48 frags * 32 lanes * 8 dwords = 48KB
    __shared__ __align__(16) uint32_t s_w[48 * 256];
    // Per-wave staging (bf16, padded pitch)
    __shared__ __align__(16) uint16_t s_q[WAVES][16 * PITCH_H];
    __shared__ __align__(16) uint16_t s_k[WAVES][TPA_T * 16 * PITCH_H];
    __shared__ __align__(16) uint16_t s_v[WAVES][TPA_T * 16 * PITCH_H];
    __shared__ __align__(16) uint32_t s_o[WAVES][16 * OPITCH];

    const int tid  = threadIdx.x;
    const int wave = tid >> 5;
    const int lane = tid & 31;
    const int lp   = lane & 15;   // N-column / pair index within tile
    const int lg   = lane >> 4;   // lane group (K-half select / M+8 select)

    // ---- one-time: swizzle all weights into per-lane B-fragment layout ----
    // frag ids: Wq: kc*4+nt (0..15) | Wk: 16+kc*4+nt (16..23)
    //           Wv: 24+kc*4+nt (24..31) | Wo: 32+kc*8+nt (32..47)
    for (int idx = tid; idx < 48 * 256; idx += blockDim.x) {
        int f = idx >> 8, rem = idx & 255, l = rem >> 3, w = rem & 7;
        const float* W; int kc, nt, pitch;
        if      (f < 16) { W = Wq; kc = f >> 2;        nt = f & 3;        pitch = 64;  }
        else if (f < 24) { W = Wk; kc = (f - 16) >> 2; nt = (f - 16) & 3; pitch = 64;  }
        else if (f < 32) { W = Wv; kc = (f - 24) >> 2; nt = (f - 24) & 3; pitch = 64;  }
        else             { W = Wo; kc = (f - 32) >> 3; nt = (f - 32) & 7; pitch = 128; }
        int krow = kc * 32 + (l >> 4) * 8 + (w >> 2) * 16 + 2 * (w & 3);   // B[K][N], table K order
        int col  = nt * 16 + (l & 15);
        float a0 = W[(size_t)krow * pitch + col];
        float a1 = W[(size_t)(krow + 1) * pitch + col];
        s_w[idx] = (uint32_t)f2bf_rne(a0) | ((uint32_t)f2bf_rne(a1) << 16);
    }
    __syncthreads();

    // per-wave constants
    float bias[TPA_T];
    #pragma unroll
    for (int tt = 0; tt < TPA_T; ++tt) bias[tt] = 1e9f * (tmask[tt] - 1.0f);
    float bo_r[8];
    #pragma unroll
    for (int nt = 0; nt < 8; ++nt) bo_r[nt] = bo[nt * 16 + lp];

    uint16_t* q_s = s_q[wave];
    uint16_t* k_s = s_k[wave];
    uint16_t* v_s = s_v[wave];
    uint32_t* o_s = s_o[wave];

    const int nwaves = gridDim.x * WAVES;
    const int kbofs  = lg * 8;   // +8 K-offset for lanes 16..31 (A layout)

    for (int tile = blockIdx.x * WAVES + wave; tile < TILES; tile += nwaves) {
        const int p0 = tile * 16;

        // ---- prefetch next tile's z (8KB) and t (4x4KB) into cache ----
        {
            int tn = tile + nwaves;
            if (tn < TILES) {
                const int pn0 = tn * 16;
                const char* zb = (const char*)(z + (size_t)pn0 * TPA_CZ);
                __builtin_prefetch(zb + lane * 256, 0, 1);
                __builtin_prefetch(zb + lane * 256 + 128, 0, 1);
                #pragma unroll
                for (int tt = 0; tt < TPA_T; ++tt) {
                    const char* tb =
                        (const char*)(tmpl + ((size_t)tt * TPA_NP + pn0) * TPA_CT);
                    __builtin_prefetch(tb + lane * 128, 0, 1);
                }
            }
        }

        // ---------- Q projection: [16 x 128] x Wq -> [16 x 64] ----------
        const float* zrow = z + (size_t)(p0 + lp) * TPA_CZ;
        Frag aq[4];
        #pragma unroll
        for (int kc = 0; kc < 4; ++kc) aq[kc] = load_afrag_global(zrow, kc * 32 + kbofs);
        #pragma unroll
        for (int nt = 0; nt < 4; ++nt) {
            v8f acc = {};
            #pragma unroll
            for (int kc = 0; kc < 4; ++kc)
                acc = wmma_bf16(aq[kc], load_bfrag(s_w, kc * 4 + nt, lane), acc);
            #pragma unroll
            for (int j = 0; j < 8; ++j)
                q_s[(j + 8 * lg) * PITCH_H + nt * 16 + lp] = f2bf(acc[j]);
        }

        // ---------- K/V projections per template (A-frags shared) ----------
        for (int tt = 0; tt < TPA_T; ++tt) {
            const float* trow = tmpl + ((size_t)tt * TPA_NP + p0 + lp) * TPA_CT;
            Frag at0 = load_afrag_global(trow, 0  + kbofs);
            Frag at1 = load_afrag_global(trow, 32 + kbofs);
            uint16_t* kb_s = k_s + tt * 16 * PITCH_H;
            uint16_t* vb_s = v_s + tt * 16 * PITCH_H;
            #pragma unroll
            for (int nt = 0; nt < 4; ++nt) {
                v8f acc = {};
                acc = wmma_bf16(at0, load_bfrag(s_w, 16 + nt, lane), acc);
                acc = wmma_bf16(at1, load_bfrag(s_w, 20 + nt, lane), acc);
                #pragma unroll
                for (int j = 0; j < 8; ++j)
                    kb_s[(j + 8 * lg) * PITCH_H + nt * 16 + lp] = f2bf(acc[j]);
            }
            #pragma unroll
            for (int nt = 0; nt < 4; ++nt) {
                v8f acc = {};
                acc = wmma_bf16(at0, load_bfrag(s_w, 24 + nt, lane), acc);
                acc = wmma_bf16(at1, load_bfrag(s_w, 28 + nt, lane), acc);
                #pragma unroll
                for (int j = 0; j < 8; ++j)
                    vb_s[(j + 8 * lg) * PITCH_H + nt * 16 + lp] = f2bf(acc[j]);
            }
        }

        // ---------- attention: lane lp = pair, lg picks 2 heads ----------
        float qv[2][16];
        #pragma unroll
        for (int hh = 0; hh < 2; ++hh)
            load16bf(q_s + lp * PITCH_H + (2 * lg + hh) * 16, qv[hh]);

        float lgt[2][TPA_T];
        #pragma unroll
        for (int tt = 0; tt < TPA_T; ++tt) {
            #pragma unroll
            for (int hh = 0; hh < 2; ++hh) {
                float kvv[16];
                load16bf(k_s + (tt * 16 + lp) * PITCH_H + (2 * lg + hh) * 16, kvv);
                float d = 0.f;
                #pragma unroll
                for (int c = 0; c < 16; ++c) d += qv[hh][c] * kvv[c];
                lgt[hh][tt] = d * 0.25f + bias[tt];   // scale = C_H^-0.5 = 1/4
            }
        }

        #pragma unroll
        for (int hh = 0; hh < 2; ++hh) {
            float mx = fmaxf(fmaxf(lgt[hh][0], lgt[hh][1]), fmaxf(lgt[hh][2], lgt[hh][3]));
            float e[TPA_T], s = 0.f;
            #pragma unroll
            for (int tt = 0; tt < TPA_T; ++tt) { e[tt] = __expf(lgt[hh][tt] - mx); s += e[tt]; }
            float inv = 1.0f / s;
            float ov[16];
            #pragma unroll
            for (int c = 0; c < 16; ++c) ov[c] = 0.f;
            #pragma unroll
            for (int tt = 0; tt < TPA_T; ++tt) {
                float a = e[tt] * inv;
                float vv[16];
                load16bf(v_s + (tt * 16 + lp) * PITCH_H + (2 * lg + hh) * 16, vv);
                #pragma unroll
                for (int c = 0; c < 16; ++c) ov[c] += a * vv[c];
            }
            uint32_t* op = o_s + lp * OPITCH + (2 * lg + hh) * 8;
            #pragma unroll
            for (int w = 0; w < 8; ++w)
                op[w] = pack2bf(ov[2 * w], ov[2 * w + 1]);
        }

        // ---------- output projection: [16 x 64] x Wo -> [16 x 128] ----------
        Frag ao[2];
        #pragma unroll
        for (int kc = 0; kc < 2; ++kc) {
            int kb = kc * 32 + kbofs;          // channel base for halves 0..7
            const uint4* pa = (const uint4*)(o_s + lp * OPITCH + kb / 2);
            const uint4* pb = (const uint4*)(o_s + lp * OPITCH + (kb + 16) / 2);
            uint4 a = pa[0], b = pb[0];
            ao[kc].u[0] = a.x; ao[kc].u[1] = a.y; ao[kc].u[2] = a.z; ao[kc].u[3] = a.w;
            ao[kc].u[4] = b.x; ao[kc].u[5] = b.y; ao[kc].u[6] = b.z; ao[kc].u[7] = b.w;
        }
        #pragma unroll
        for (int nt = 0; nt < 8; ++nt) {
            v8f acc = {};
            acc = wmma_bf16(ao[0], load_bfrag(s_w, 32 + nt, lane), acc);
            acc = wmma_bf16(ao[1], load_bfrag(s_w, 40 + nt, lane), acc);
            #pragma unroll
            for (int j = 0; j < 8; ++j)
                out[(size_t)(p0 + j + 8 * lg) * TPA_CZ + nt * 16 + lp] = acc[j] + bo_r[nt];
        }
    }
}

extern "C" void kernel_launch(void* const* d_in, const int* in_sizes, int n_in,
                              void* d_out, int out_size, void* d_ws, size_t ws_size,
                              hipStream_t stream) {
    (void)in_sizes; (void)n_in; (void)out_size; (void)d_ws; (void)ws_size;
    const float* t   = (const float*)d_in[0];
    const float* z   = (const float*)d_in[1];
    const float* msk = (const float*)d_in[2];
    const float* Wq  = (const float*)d_in[3];
    const float* Wk  = (const float*)d_in[4];
    const float* Wv  = (const float*)d_in[5];
    const float* Wo  = (const float*)d_in[6];
    const float* bo  = (const float*)d_in[7];
    float* out = (float*)d_out;

    dim3 grid(1024), block(WAVES * 32);
    tpa_wmma_kernel<<<grid, block, 0, stream>>>(t, z, msk, Wq, Wk, Wv, Wo, bo, out);
}
